// SubtypeAwareMWCL_22299470201109
// MI455X (gfx1250) — compile-verified, matrix operand order checked by reading, not run on hardware
//
#include <hip/hip_runtime.h>
#include <hip/hip_bf16.h>
#include <math.h>

typedef _Float16 v16h __attribute__((ext_vector_type(16)));
typedef _Float16 v8h  __attribute__((ext_vector_type(8)));
typedef _Float16 v4h  __attribute__((ext_vector_type(4)));
typedef float    v8f  __attribute__((ext_vector_type(8)));

#define NB 4096
#define ND 1024
#define LDA 1032              // padded row stride (halves): 2064B = 516 dwords -> conflict-free 16-lane b128
#define INV_T 14.285714285714286f
#define NT 2                  // column tiles per wave per iteration

// ---------------------------------------------------------------- zero output
__global__ __launch_bounds__(32) void k_zero(float* out) {
    if (threadIdx.x == 0) out[0] = 0.0f;
}

// ------------------------------------------- row-normalize + convert to f16
__global__ __launch_bounds__(256) void k_norm(const float* __restrict__ M,
                                              const float* __restrict__ P,
                                              _Float16* __restrict__ Mh,
                                              _Float16* __restrict__ Ph) {
    int bid = blockIdx.x;
    const float* src = (bid < NB) ? M : P;
    _Float16*    dst = (bid < NB) ? Mh : Ph;
    int row = (bid < NB) ? bid : bid - NB;

    const float4* s4 = (const float4*)(src + (size_t)row * ND);
    float4 v = s4[threadIdx.x];                       // 256 threads * 4 = 1024
    float ss = v.x * v.x + v.y * v.y + v.z * v.z + v.w * v.w;
    #pragma unroll
    for (int m = 16; m >= 1; m >>= 1) ss += __shfl_xor(ss, m, 32);

    __shared__ float buf[8];
    int lane = threadIdx.x & 31, wv = threadIdx.x >> 5;
    if (lane == 0) buf[wv] = ss;
    __syncthreads();
    float tot = 0.f;
    #pragma unroll
    for (int w = 0; w < 8; ++w) tot += buf[w];
    float inv = 1.0f / sqrtf(tot);

    v4h o;
    o.x = (_Float16)(v.x * inv); o.y = (_Float16)(v.y * inv);
    o.z = (_Float16)(v.z * inv); o.w = (_Float16)(v.w * inv);
    *(v4h*)(dst + (size_t)row * ND + threadIdx.x * 4) = o;
}

// --------------------------------------------- per-row weight denominators
__global__ __launch_bounds__(256) void k_rowsum(const int* __restrict__ lab,
                                                const int* __restrict__ sm,
                                                const int* __restrict__ sp,
                                                const float* __restrict__ clin,
                                                float* __restrict__ rowsum) {
    int i = blockIdx.x;
    int li = lab[i], mi = sm[i], pi = sp[i];
    float s = 0.f;
    for (int j = threadIdx.x; j < NB; j += 256) {
        bool msk = (lab[j] == li) &&
                   ((li == 0) || ((sm[j] == mi) && (sp[j] == pi))) &&
                   (j != i);
        if (msk) s += clin[(size_t)i * NB + j];
    }
    #pragma unroll
    for (int m = 16; m >= 1; m >>= 1) s += __shfl_xor(s, m, 32);
    __shared__ float buf[8];
    int lane = threadIdx.x & 31, wv = threadIdx.x >> 5;
    if (lane == 0) buf[wv] = s;
    __syncthreads();
    if (threadIdx.x == 0) {
        float t = 0.f;
        #pragma unroll
        for (int w = 0; w < 8; ++w) t += buf[w];
        rowsum[i] = t;
    }
}

// --------------------- fused GEMM (WMMA f16) + log-softmax-weighted reduce
// block = 8 waves, one 16-row strip of one (fi,fj) pair; waves sweep column
// tiles NT at a time with double-buffered fragment loads. The empty asm on
// aOff makes the A-fragment LDS address opaque per tg-iteration, preventing
// the compiler from hoisting all 32 k-steps of A fragments (256 VGPRs) out
// of the loop (which caused VGPR-MSB spam in one build, scratch spills in
// another). Live set stays ~130 VGPRs: no spills, good occupancy.
__global__ __launch_bounds__(256)
void k_main(const _Float16* __restrict__ Mh,
            const _Float16* __restrict__ Ph,
            const int* __restrict__ lab,
            const int* __restrict__ sm,
            const int* __restrict__ sp,
            const float* __restrict__ clin,
            const float* __restrict__ rowsum,
            float* __restrict__ out) {
    __shared__ _Float16 sA[16 * LDA];
    __shared__ float sExp[16], sW[16], sC[16];

    const int bid  = blockIdx.x;
    const int pair = bid >> 8;              // 0:(M,P) 1:(P,M) 2:(M,M) 3:(P,P)
    const int r0   = (bid & 255) << 4;
    const _Float16* fi = (pair == 0 || pair == 2) ? Mh : Ph;
    const _Float16* fj = (pair == 1 || pair == 2) ? Mh : Ph;
    const char* fjB = (const char*)fj;      // uniform base + u32 byte offsets

    const int tid  = threadIdx.x;
    const int lane = tid & 31;
    const int wv   = tid >> 5;
    const int half = lane >> 4;             // 0 or 1
    const int lcol = lane & 15;

    if (tid < 16) { sExp[tid] = 0.f; sW[tid] = 0.f; }

    // stage A strip (16 rows x 1024 halves) into padded LDS
    for (int c = tid; c < 2048; c += 256) {
        int row = c >> 7;
        int kc  = (c & 127) << 3;
        *(v8h*)(&sA[row * LDA + kc]) =
            *(const v8h*)(fi + (size_t)(r0 + row) * ND + kc);
    }
    __syncthreads();

    // row-side metadata, packed: key equality implies subtype+label equality
    int labi[8], keyi[8];
    #pragma unroll
    for (int r = 0; r < 8; ++r) {
        int i = r0 + half * 8 + r;
        labi[r] = lab[i];
        keyi[r] = (lab[i] << 20) | (sm[i] << 10) | sp[i];
    }

    float accE[8], accW[8];
    #pragma unroll
    for (int r = 0; r < 8; ++r) { accE[r] = 0.f; accW[r] = 0.f; }

    const char* sAB = (const char*)sA;      // LDS base as bytes
    const unsigned aOff0 = (unsigned)(lcol * LDA + half * 8) * 2u;

    for (int tg = wv; tg < 256 / NT; tg += 8) {
        // opaque per-iteration A offset: blocks loop-invariant hoisting of
        // the A-fragment ds_loads (zero instructions emitted)
        unsigned aOff = aOff0;
        asm volatile("" : "+v"(aOff));

        int labj[NT], keyj[NT];
        unsigned boff[NT];                  // byte offset of this lane's B row
        #pragma unroll
        for (int tt = 0; tt < NT; ++tt) {
            int j    = ((tg * NT + tt) << 4) + lcol;
            labj[tt] = lab[j];
            keyj[tt] = (lab[j] << 20) | (sm[j] << 10) | sp[j];
            boff[tt] = (unsigned)j * (ND * 2u) + (unsigned)half * 16u;
        }

        v8f c[NT] = {};

        // ---- software pipeline: prologue loads for k = 0
        v16h a_cur;
        ((v8h*)&a_cur)[0] = *(const v8h*)(sAB + aOff);
        ((v8h*)&a_cur)[1] = *(const v8h*)(sAB + aOff + 32u);
        v16h b_cur[NT];
        #pragma unroll
        for (int tt = 0; tt < NT; ++tt) {
            ((v8h*)&b_cur[tt])[0] = *(const v8h*)(fjB + boff[tt]);
            ((v8h*)&b_cur[tt])[1] = *(const v8h*)(fjB + boff[tt] + 32u);
        }

        #pragma unroll 4
        for (int k0 = 32; k0 < ND; k0 += 32) {
            // issue next iteration's loads first
            v16h a_nxt, b_nxt[NT];
            unsigned ao = aOff + (unsigned)(2 * k0);
            ((v8h*)&a_nxt)[0] = *(const v8h*)(sAB + ao);
            ((v8h*)&a_nxt)[1] = *(const v8h*)(sAB + ao + 32u);
            #pragma unroll
            for (int tt = 0; tt < NT; ++tt) {
                unsigned o = boff[tt] + (unsigned)(2 * k0);
                ((v8h*)&b_nxt[tt])[0] = *(const v8h*)(fjB + o);
                ((v8h*)&b_nxt[tt])[1] = *(const v8h*)(fjB + o + 32u);
            }
            // consume current buffers (loads above stay in flight)
            #pragma unroll
            for (int tt = 0; tt < NT; ++tt)
                c[tt] = __builtin_amdgcn_wmma_f32_16x16x32_f16(
                    false, a_cur, false, b_cur[tt], (short)0, c[tt], false, false);
            a_cur = a_nxt;
            #pragma unroll
            for (int tt = 0; tt < NT; ++tt) b_cur[tt] = b_nxt[tt];
        }
        // ---- drain: last k-step
        #pragma unroll
        for (int tt = 0; tt < NT; ++tt)
            c[tt] = __builtin_amdgcn_wmma_f32_16x16x32_f16(
                false, a_cur, false, b_cur[tt], (short)0, c[tt], false, false);

        // epilogue: softmax stats + weighted sim, element (i = r0+half*8+r, j)
        #pragma unroll
        for (int tt = 0; tt < NT; ++tt) {
            int j = ((tg * NT + tt) << 4) + lcol;
            #pragma unroll
            for (int r = 0; r < 8; ++r) {
                int i   = r0 + half * 8 + r;
                float s = c[tt][r] * INV_T;
                accE[r] += __expf(s);   // no max needed: s <= 14.3, sum <= 6.7e9
                bool msk = (labi[r] == labj[tt]) &&
                           ((labi[r] == 0) || (keyi[r] == keyj[tt])) &&
                           (i != j);
                if (msk) accW[r] += clin[(size_t)i * NB + j] * s;
            }
        }
    }

    // reduce across the 16 lanes of each half (columns of the tile row)
    #pragma unroll
    for (int r = 0; r < 8; ++r) {
        float e = accE[r], w = accW[r];
        #pragma unroll
        for (int m = 1; m < 16; m <<= 1) {
            e += __shfl_xor(e, m, 32);
            w += __shfl_xor(w, m, 32);
        }
        if (lcol == 0) {
            atomicAdd(&sExp[half * 8 + r], e);
            atomicAdd(&sW[half * 8 + r], w);
        }
    }
    __syncthreads();

    if (tid < 16) {
        float rs = rowsum[r0 + tid];
        float contrib = 0.f;
        if (rs > 0.f) contrib = __logf(sExp[tid]) - sW[tid] / rs;
        sC[tid] = contrib;
    }
    __syncthreads();
    if (tid == 0) {
        float t = 0.f;
        #pragma unroll
        for (int k = 0; k < 16; ++k) t += sC[k];
        atomicAdd(out, t * (1.0f / (4.0f * (float)NB)));
    }
}

// ---------------------------------------------------------------- launcher
extern "C" void kernel_launch(void* const* d_in, const int* in_sizes, int n_in,
                              void* d_out, int out_size, void* d_ws, size_t ws_size,
                              hipStream_t stream) {
    const float* M    = (const float*)d_in[0];
    const float* P    = (const float*)d_in[1];
    const int*   lab  = (const int*)d_in[2];
    const int*   smv  = (const int*)d_in[3];
    const int*   spv  = (const int*)d_in[4];
    const float* clin = (const float*)d_in[5];
    float* out = (float*)d_out;

    // workspace: Mh (8MB) | Ph (8MB) | rowsum (16KB)
    _Float16* Mh = (_Float16*)d_ws;
    _Float16* Ph = Mh + (size_t)NB * ND;
    float* rowsum = (float*)(Ph + (size_t)NB * ND);

    k_zero  <<<1,       32, 0, stream>>>(out);
    k_norm  <<<2 * NB, 256, 0, stream>>>(M, P, Mh, Ph);
    k_rowsum<<<NB,     256, 0, stream>>>(lab, smv, spv, clin, rowsum);
    k_main  <<<4 * 256, 256, 0, stream>>>(Mh, Ph, lab, smv, spv, clin, rowsum, out);
}
